// NonLocalNet_9328668967365
// MI455X (gfx1250) — matrix operand order repaired and checked by reading, not compiled, not run on hardware
//
#include <hip/hip_runtime.h>

#define B_    8192
#define C_    1024
#define P_    8
#define DCUT_ 128
#define UP_   1024
#define OUT_  512
#define BT    32          // batch rows per workgroup = two 16-row WMMA M-tiles

#define EMB_ATT_N (B_ * P_ * DCUT_)      // 8,388,608 bf16
#define WPS_N     (P_ * DCUT_ * C_)      // 1,048,576
#define WUP_N     (P_ * UP_ * DCUT_)     // 1,048,576
#define WLA_N     (P_ * OUT_ * C_)       // 4,194,304

typedef __attribute__((ext_vector_type(16))) __bf16 v16bf;
typedef __attribute__((ext_vector_type(8)))  float  v8f;

// ---- fragment loaders -------------------------------------------------------
// A (16x32 bf16): lane m = lane&15, half h = lane>>4.
// element e -> k = (e&7) + 8*h + 16*(e>>3). base = &row[k0 + 8*h] (16B aligned)
__device__ __forceinline__ v16bf ld_a_lds(const __bf16* base) {
  union { uint4 u[2]; v16bf v; } t;
  t.u[0] = *reinterpret_cast<const uint4*>(base);
  t.u[1] = *reinterpret_cast<const uint4*>(base + 16);
  return t.v;
}

// B (32x16 bf16): lane n = lane&15, element e -> k = (e&7) + 8*h + 16*(e>>3).
// Pre-converted bf16 weights, row-major over K: two 8-bf16 (16B) runs at base and base+16.
__device__ __forceinline__ v16bf ld_b_bf16(const __bf16* base) {
  union { uint4 u[2]; v16bf v; } t;
  t.u[0] = *reinterpret_cast<const uint4*>(base);
  t.u[1] = *reinterpret_cast<const uint4*>(base + 16);
  return t.v;
}

__device__ __forceinline__ v8f wmma_bf16(v16bf a, v16bf b, v8f c) {
  return __builtin_amdgcn_wmma_f32_16x16x32_bf16(false, a, false, b,
                                                 (short)0, c, false, false);
}

__device__ __forceinline__ float lrelu(float x) { return x > 0.0f ? x : 0.25f * x; }

// ---- Weight pre-conversion: f32 -> bf16 (once per launch, trivially BW-bound)
__global__ __launch_bounds__(256) void cvt_f32_bf16(const float* __restrict__ s,
                                                    __bf16* __restrict__ d, int n4) {
  const int i = blockIdx.x * 256 + threadIdx.x;
  if (i < n4) {
    const float4 v = reinterpret_cast<const float4*>(s)[i];
    union { __bf16 h[4]; uint2 u; } t;
    t.h[0] = (__bf16)v.x; t.h[1] = (__bf16)v.y;
    t.h[2] = (__bf16)v.z; t.h[3] = (__bf16)v.w;
    reinterpret_cast<uint2*>(d)[i] = t.u;
  }
}

// ---- Kernel 1: ps/cs GEMMs + l2norm + attention + softmax -> emb_att (bf16) -
// Parts in pairs (float2 staging), 8 WMMAs/k-step, B double-buffered across
// k-steps, embedding chunks double-buffered through registers.
__global__ __launch_bounds__(256) void k1_att(const float* __restrict__ emb,
                                              const __bf16* __restrict__ Wps,
                                              const __bf16* __restrict__ Wcs,
                                              __bf16* __restrict__ emb_att) {
  __shared__ __bf16 es[2][BT][DCUT_ + 8];         // emb K-chunk, 2 parts (bf16)
  __shared__ __bf16 ps_s[BT][P_][DCUT_ + 2];      // lrelu(ps), bf16
  __shared__ __bf16 cs_s[BT][P_][DCUT_ + 2];      // lrelu(cs), bf16
  __shared__ float  inv_p[BT][P_];
  __shared__ float  inv_c[BT][P_];
  __shared__ float  att_s[BT][P_][P_];

  const int tid  = threadIdx.x;
  const int lane = tid & 31;
  const int wv   = tid >> 5;          // 8 waves: one 16-wide D-tile each
  const int hh   = lane >> 4;
  const int ln   = lane & 15;
  const long b0  = (long)blockIdx.x * BT;

  for (int p = 0; p < P_; p += 2) {
    v8f accp[2][2] = {};              // [part-in-pair][M-tile]
    v8f accc[2][2] = {};
    const int d0 = wv * 16;
    const __bf16* wp0 = Wps + ((long)p * DCUT_ + (d0 + ln)) * C_ + 8 * hh;
    const __bf16* wc0 = Wcs + ((long)p * DCUT_ + (d0 + ln)) * C_ + 8 * hh;
    const __bf16* wp1 = wp0 + (long)DCUT_ * C_;   // same row, part p+1
    const __bf16* wc1 = wc0 + (long)DCUT_ * C_;

    // stage chunk 0
    __syncthreads();
    for (int i = tid; i < BT * DCUT_; i += 256) {
      const int m = i >> 7, k = i & (DCUT_ - 1);
      const float2 v = *reinterpret_cast<const float2*>(
          emb + ((b0 + m) * C_ + k) * P_ + p);
      es[0][m][k] = (__bf16)v.x;
      es[1][m][k] = (__bf16)v.y;
    }
    __syncthreads();

    // B fragments for kk = 0
    v16bf bpn0 = ld_b_bf16(wp0), bcn0 = ld_b_bf16(wc0);
    v16bf bpn1 = ld_b_bf16(wp1), bcn1 = ld_b_bf16(wc1);

    for (int kci = 0; kci < 8; ++kci) {
      const int kc = kci * DCUT_;
      // Issue next chunk's gathers now; latency hidden under the 32 WMMAs below.
      float2 sreg[16];
      if (kci < 7) {
        #pragma unroll
        for (int j = 0; j < 16; ++j) {
          const int i = tid + j * 256;
          const int m = i >> 7, k = i & (DCUT_ - 1);
          sreg[j] = *reinterpret_cast<const float2*>(
              emb + ((b0 + m) * C_ + (kc + DCUT_ + k)) * P_ + p);
        }
      }
      #pragma unroll
      for (int k0 = 0; k0 < DCUT_; k0 += 32) {
        const int kk = kc + k0;
        v16bf bp0 = bpn0, bc0 = bcn0, bp1 = bpn1, bc1 = bcn1;
        // Unconditional next-step prefetch. Final step over-reads <=64B past a
        // row, landing in the next (mapped) bf16 workspace region: harmless.
        bpn0 = ld_b_bf16(wp0 + kk + 32); bcn0 = ld_b_bf16(wc0 + kk + 32);
        bpn1 = ld_b_bf16(wp1 + kk + 32); bcn1 = ld_b_bf16(wc1 + kk + 32);
        v16bf a00 = ld_a_lds(&es[0][ln][k0 + 8 * hh]);
        v16bf a01 = ld_a_lds(&es[0][16 + ln][k0 + 8 * hh]);
        v16bf a10 = ld_a_lds(&es[1][ln][k0 + 8 * hh]);
        v16bf a11 = ld_a_lds(&es[1][16 + ln][k0 + 8 * hh]);
        accp[0][0] = wmma_bf16(a00, bp0, accp[0][0]);
        accp[0][1] = wmma_bf16(a01, bp0, accp[0][1]);
        accc[0][0] = wmma_bf16(a00, bc0, accc[0][0]);
        accc[0][1] = wmma_bf16(a01, bc0, accc[0][1]);
        accp[1][0] = wmma_bf16(a10, bp1, accp[1][0]);
        accp[1][1] = wmma_bf16(a11, bp1, accp[1][1]);
        accc[1][0] = wmma_bf16(a10, bc1, accc[1][0]);
        accc[1][1] = wmma_bf16(a11, bc1, accc[1][1]);
      }
      __syncthreads();                 // all reads of es for this chunk done
      if (kci < 7) {
        #pragma unroll
        for (int j = 0; j < 16; ++j) {
          const int i = tid + j * 256;
          const int m = i >> 7, k = i & (DCUT_ - 1);
          es[0][m][k] = (__bf16)sreg[j].x;
          es[1][m][k] = (__bf16)sreg[j].y;
        }
      }
      __syncthreads();                 // es[next] visible to all waves
    }

    #pragma unroll
    for (int pp = 0; pp < 2; ++pp) {
      #pragma unroll
      for (int r = 0; r < 8; ++r) {               // D frag: m = r + 8h, n = ln
        const int m = r + 8 * hh;
        const int d = d0 + ln;
        ps_s[m][p + pp][d]      = (__bf16)lrelu(accp[pp][0][r]);
        ps_s[m + 16][p + pp][d] = (__bf16)lrelu(accp[pp][1][r]);
        cs_s[m][p + pp][d]      = (__bf16)lrelu(accc[pp][0][r]);
        cs_s[m + 16][p + pp][d] = (__bf16)lrelu(accc[pp][1][r]);
      }
    }
  }
  __syncthreads();

  // l2 norms over d: BT*P_ == 256 == blockDim -> no divergence
  {
    const int m = tid >> 3, p = tid & 7;
    float sp = 0.f, sc = 0.f;
    for (int d = 0; d < DCUT_; ++d) {
      const float x = (float)ps_s[m][p][d]; sp += x * x;
      const float y = (float)cs_s[m][p][d]; sc += y * y;
    }
    inv_p[m][p] = 1.0f / (sqrtf(sp) + 1e-8f);
    inv_c[m][p] = 1.0f / (sqrtf(sc) + 1e-8f);
  }
  __syncthreads();

  // att[b,pp,q] = <psn[:,pp], csn[:,q]> + diagonal mask
  for (int idx = tid; idx < BT * P_ * P_; idx += 256) {
    const int m = idx >> 6, pp = (idx >> 3) & 7, q = idx & 7;
    float dot = 0.f;
    for (int d = 0; d < DCUT_; ++d)
      dot += (float)ps_s[m][pp][d] * (float)cs_s[m][q][d];
    float v = dot * inv_p[m][pp] * inv_c[m][q];
    if (pp == q) v += -1.0e6f;
    att_s[m][pp][q] = v;
  }
  __syncthreads();

  // softmax over pp (axis=1); BT*P_ == 256
  {
    const int m = tid >> 3, q = tid & 7;
    float mx = -3.0e38f;
    for (int pp = 0; pp < P_; ++pp) mx = fmaxf(mx, att_s[m][pp][q]);
    float e[P_], s = 0.f;
    for (int pp = 0; pp < P_; ++pp) { e[pp] = __expf(att_s[m][pp][q] - mx); s += e[pp]; }
    const float is = 1.0f / s;
    for (int pp = 0; pp < P_; ++pp) att_s[m][pp][q] = e[pp] * is;
  }
  __syncthreads();

  // emb_att[b,d,q] = sum_pp psn[b,d,pp] * att[b,pp,q]; store [b][q][d] bf16
  for (int idx = tid; idx < BT * P_ * DCUT_; idx += 256) {
    const int d = idx & (DCUT_ - 1);
    const int q = (idx >> 7) & 7;
    const int m = idx >> 10;
    float s = 0.f;
    #pragma unroll
    for (int pp = 0; pp < P_; ++pp)
      s += (float)ps_s[m][pp][d] * inv_p[m][pp] * att_s[m][pp][q];
    emb_att[((b0 + m) * P_ + q) * DCUT_ + d] = (__bf16)s;
  }
}

// ---- Kernel 2: up GEMM + BN + lrelu + residual + la GEMM -> out -------------
__global__ __launch_bounds__(256) void k2_out(const float* __restrict__ emb,
                                              const __bf16* __restrict__ emb_att,
                                              const __bf16* __restrict__ Wup,
                                              const float* __restrict__ gamma,
                                              const float* __restrict__ beta,
                                              const __bf16* __restrict__ Wla,
                                              float* __restrict__ out) {
  __shared__ __bf16 a1s[BT][DCUT_ + 8];           // emb_att tile
  __shared__ __bf16 rs[BT][C_ + 8];               // residual tile
  const int tid  = threadIdx.x;
  const int lane = tid & 31, wv = tid >> 5;
  const int hh   = lane >> 4, ln = lane & 15;
  const long b0  = (long)blockIdx.x * BT;
  const int p    = blockIdx.y;

  for (int i = tid; i < BT * DCUT_; i += 256) {
    const int m = i >> 7, d = i & (DCUT_ - 1);
    a1s[m][d] = emb_att[((b0 + m) * P_ + p) * DCUT_ + d];
  }
  __syncthreads();

  const float bninv = rsqrtf(1.0f + 1.0e-5f);
  // Stage 1: up = emb_att @ Wup[p]^T; per wave 8 o-tiles, 2 M-tiles each.
  // B double-buffered; over-read past the last row lands in Wla copy (mapped).
  for (int jt = 0; jt < 8; ++jt) {
    const int o0 = wv * 128 + jt * 16;
    v8f acc0 = {}, acc1 = {};
    const __bf16* wrow = Wup + ((long)p * UP_ + (o0 + ln)) * DCUT_ + 8 * hh;
    v16bf bn = ld_b_bf16(wrow);
    #pragma unroll
    for (int k0 = 0; k0 < DCUT_; k0 += 32) {
      v16bf b = bn;
      bn = ld_b_bf16(wrow + k0 + 32);              // last step over-read: benign
      v16bf a0 = ld_a_lds(&a1s[ln][k0 + 8 * hh]);
      v16bf a1 = ld_a_lds(&a1s[16 + ln][k0 + 8 * hh]);
      acc0 = wmma_bf16(a0, b, acc0);
      acc1 = wmma_bf16(a1, b, acc1);
    }
    const int o = o0 + ln;
    const float g  = gamma[p * UP_ + o] * bninv;
    const float bt = beta[p * UP_ + o];
    #pragma unroll
    for (int r = 0; r < 8; ++r) {
      const int m = r + 8 * hh;
      float v0 = lrelu(acc0[r] * g + bt) + emb[((b0 + m) * C_ + o) * P_ + p];
      float v1 = lrelu(acc1[r] * g + bt) + emb[((b0 + 16 + m) * C_ + o) * P_ + p];
      rs[m][o]      = (__bf16)v0;                  // residual (UP == C)
      rs[m + 16][o] = (__bf16)v1;
    }
  }
  __syncthreads();

  // Stage 2: out = res @ Wla[p]^T; per wave 4 o2-tiles, 2 M-tiles each.
  // B double-buffered; last iteration peeled (Wla is last ws region, no over-read).
  for (int jt = 0; jt < 4; ++jt) {
    const int o0 = (wv * 4 + jt) * 16;
    v8f acc0 = {}, acc1 = {};
    const __bf16* wrow = Wla + ((long)p * OUT_ + (o0 + ln)) * C_ + 8 * hh;
    v16bf bcur = ld_b_bf16(wrow);
    #pragma unroll 4
    for (int k0 = 0; k0 < C_ - 32; k0 += 32) {
      v16bf b = bcur;
      bcur = ld_b_bf16(wrow + k0 + 32);
      v16bf a0 = ld_a_lds(&rs[ln][k0 + 8 * hh]);
      v16bf a1 = ld_a_lds(&rs[16 + ln][k0 + 8 * hh]);
      acc0 = wmma_bf16(a0, b, acc0);
      acc1 = wmma_bf16(a1, b, acc1);
    }
    {
      const int k0 = C_ - 32;
      v16bf a0 = ld_a_lds(&rs[ln][k0 + 8 * hh]);
      v16bf a1 = ld_a_lds(&rs[16 + ln][k0 + 8 * hh]);
      acc0 = wmma_bf16(a0, bcur, acc0);
      acc1 = wmma_bf16(a1, bcur, acc1);
    }
    const int o2 = o0 + ln;
    #pragma unroll
    for (int r = 0; r < 8; ++r) {
      const int m = r + 8 * hh;
      out[((b0 + m) * OUT_ + o2) * P_ + p]      = acc0[r];
      out[((b0 + 16 + m) * OUT_ + o2) * P_ + p] = acc1[r];
    }
  }
}

extern "C" void kernel_launch(void* const* d_in, const int* in_sizes, int n_in,
                              void* d_out, int out_size, void* d_ws, size_t ws_size,
                              hipStream_t stream) {
  (void)in_sizes; (void)n_in; (void)out_size; (void)ws_size;
  const float* emb   = (const float*)d_in[0];
  const float* Wps   = (const float*)d_in[1];
  const float* Wcs   = (const float*)d_in[2];
  const float* Wup   = (const float*)d_in[3];
  const float* gamma = (const float*)d_in[4];
  const float* beta  = (const float*)d_in[5];
  const float* Wla   = (const float*)d_in[6];
  float* out = (float*)d_out;

  // Workspace layout (30 MB total): emb_att bf16, then bf16 weight copies.
  // Region order matters: benign 64B prefetch over-reads fall into the next region.
  __bf16* emb_att = (__bf16*)d_ws;                 // 16 MB
  __bf16* wps_bf  = emb_att + EMB_ATT_N;           // 2 MB
  __bf16* wcs_bf  = wps_bf + WPS_N;                // 2 MB
  __bf16* wup_bf  = wcs_bf + WPS_N;                // 2 MB
  __bf16* wla_bf  = wup_bf + WUP_N;                // 8 MB

  cvt_f32_bf16<<<WPS_N / 4 / 256, 256, 0, stream>>>(Wps, wps_bf, WPS_N / 4);
  cvt_f32_bf16<<<WPS_N / 4 / 256, 256, 0, stream>>>(Wcs, wcs_bf, WPS_N / 4);
  cvt_f32_bf16<<<WUP_N / 4 / 256, 256, 0, stream>>>(Wup, wup_bf, WUP_N / 4);
  cvt_f32_bf16<<<WLA_N / 4 / 256, 256, 0, stream>>>(Wla, wla_bf, WLA_N / 4);

  k1_att<<<dim3(B_ / BT), 256, 0, stream>>>(emb, wps_bf, wcs_bf, emb_att);
  k2_out<<<dim3(B_ / BT, P_), 256, 0, stream>>>(emb, emb_att, wup_bf, gamma,
                                                beta, wla_bf, out);
}
